// MaskedCrossAttention_11416023073046
// MI455X (gfx1250) — compile-verified, matrix operand order checked by reading, not run on hardware
//
#include <hip/hip_runtime.h>

typedef __attribute__((ext_vector_type(16))) _Float16 v16h;
typedef __attribute__((ext_vector_type(8)))  _Float16 v8h;
typedef __attribute__((ext_vector_type(4)))  _Float16 v4h;
typedef __attribute__((ext_vector_type(8)))  float    v8f;

#define DIM_   1024
#define INNER_ 512
#define HEADS_ 8
#define DHEAD_ 64
#define TSEQ_  2048
#define BATCH_ 4
#define ROWS_  (BATCH_*TSEQ_)   // 8192

// ---------------------------------------------------------------------------
// Fragment load helper. Per CDNA5 ISA (CDNA5 ISA, 16-bit A 16x32 layout):
// lanes 0-15 hold K in [c0, c0+8) u [c0+16, c0+24); lanes 16-31 the mirrored
// halves (c0 = hw*8). Caller passes p = base + hw*8; we read 2x16B.
// Same interleave for B (column-per-lane, K-contiguous chunks).
// ---------------------------------------------------------------------------
__device__ __forceinline__ v16h frag_ld(const _Float16* p) {
  v8h lo = *(const v8h*)p;
  v8h hi = *(const v8h*)(p + 16);
  return __builtin_shufflevector(lo, hi, 0,1,2,3,4,5,6,7,8,9,10,11,12,13,14,15);
}

__device__ __forceinline__ v8f wmma16(v16h a, v16h b, v8f c) {
  // v_wmma_f32_16x16x32_f16 (neg_a, A, neg_b, B, c_mod, C, reuse_a, reuse_b)
  return __builtin_amdgcn_wmma_f32_16x16x32_f16(false, a, false, b, (short)0, c, false, false);
}

// ---------------------------------------------------------------------------
// CDNA5 async copy: global -> LDS, 16B per lane, tracked with ASYNCcnt.
// (CDNA5 ISA op 98 GLOBAL_LOAD_ASYNC_TO_LDS_B128, GV mode.)
// LDS address = low 32 bits of the generic pointer (ISA aperture rule:
// applies to LDS-aperture flat addresses).
// ---------------------------------------------------------------------------
__device__ __forceinline__ void async_copy_b128(_Float16* ldst, const _Float16* gsrc) {
  unsigned loff = (unsigned)(unsigned long long)ldst;
  asm volatile("global_load_async_to_lds_b128 %0, %1, off"
               :: "v"(loff), "v"(gsrc)
               : "memory");
}
__device__ __forceinline__ void wait_async1() {
  asm volatile("s_wait_asynccnt 0x1" ::: "memory");
}
__device__ __forceinline__ void wait_async0() {
  asm volatile("s_wait_asynccnt 0x0" ::: "memory");
}

// ---------------------------------------------------------------------------
// Weight transpose + f32->f16:  dst[n*K + k] = src[k*ld + off + n]
// ---------------------------------------------------------------------------
__global__ __launch_bounds__(256) void wt_trans(const float* __restrict__ src,
                                                _Float16* __restrict__ dst,
                                                int K, int N, int ld, int off) {
  int idx = blockIdx.x * 256 + threadIdx.x;
  if (idx >= K * N) return;
  int n = idx / K;
  int k = idx - n * K;
  dst[idx] = (_Float16)src[(size_t)k * ld + off + n];
}

// ---------------------------------------------------------------------------
// LayerNorm over DIM_=1024 + f32->f16.  One 256-thread block per row.
// ---------------------------------------------------------------------------
__global__ __launch_bounds__(256) void ln_f16(const float* __restrict__ x,
                                              const float* __restrict__ sc,
                                              const float* __restrict__ bi,
                                              _Float16* __restrict__ out) {
  const int row = blockIdx.x;
  const int tid = threadIdx.x;
  const float4 v = ((const float4*)(x + (size_t)row * DIM_))[tid];
  float s = v.x + v.y + v.z + v.w;
  float q = v.x * v.x + v.y * v.y + v.z * v.z + v.w * v.w;
  for (int off = 1; off < 32; off <<= 1) {
    s += __shfl_xor(s, off, 32);
    q += __shfl_xor(q, off, 32);
  }
  __shared__ float ss[8], sq[8];
  if ((tid & 31) == 0) { ss[tid >> 5] = s; sq[tid >> 5] = q; }
  __syncthreads();
  float ts = 0.f, tq = 0.f;
  for (int i = 0; i < 8; ++i) { ts += ss[i]; tq += sq[i]; }
  const float mean = ts * (1.f / DIM_);
  const float var  = tq * (1.f / DIM_) - mean * mean;
  const float rstd = rsqrtf(var + 1e-5f);
  const float4 scv = ((const float4*)sc)[tid];
  const float4 biv = ((const float4*)bi)[tid];
  v4h o;
  o[0] = (_Float16)((v.x - mean) * rstd * scv.x + biv.x);
  o[1] = (_Float16)((v.y - mean) * rstd * scv.y + biv.y);
  o[2] = (_Float16)((v.z - mean) * rstd * scv.z + biv.z);
  o[3] = (_Float16)((v.w - mean) * rstd * scv.w + biv.w);
  ((v4h*)(out + (size_t)row * DIM_))[tid] = o;
}

// ---------------------------------------------------------------------------
// f32 -> f16 elementwise (float4 granularity)
// ---------------------------------------------------------------------------
__global__ __launch_bounds__(256) void conv_f16(const float* __restrict__ in,
                                                _Float16* __restrict__ out, int n4) {
  int i = blockIdx.x * 256 + threadIdx.x;
  if (i >= n4) return;
  float4 v = ((const float4*)in)[i];
  v4h o;
  o[0] = (_Float16)v.x; o[1] = (_Float16)v.y;
  o[2] = (_Float16)v.z; o[3] = (_Float16)v.w;
  ((v4h*)out)[i] = o;
}

// ---------------------------------------------------------------------------
// WMMA GEMM:  C[M,N] = scale * A[M,K] @ WT[N,K]^T
// Block = 256 thr = 8 waves. Wave computes 16(M) x 64(N). Grid (M/128, N/64).
// The 64x32 B tile (shared by all 8 waves) is double-buffered in LDS via
// global_load_async_to_lds_b128: each wave issues one async b128/K-step
// (256 lanes x 16B = 4KB tile), waits own ASYNCcnt, then block barrier.
// MODE 0: f16 row-major C; MODE 1: f16 transposed per-batch into
//         vT[(b*N + n)*TSEQ_ + t] (contiguous v8h along M); MODE 2: f32 C.
// ---------------------------------------------------------------------------
template <int MODE>
__global__ __launch_bounds__(256) void gemm_wmma(const _Float16* __restrict__ A,
                                                 const _Float16* __restrict__ WT,
                                                 void* __restrict__ Cout,
                                                 int M, int N, int K, float scale) {
  const int tid = threadIdx.x;
  const int l  = tid & 31;
  const int w  = tid >> 5;
  const int ln = l & 15;
  const int hw = l >> 4;
  const int m0 = blockIdx.x * 128 + w * 16;
  const int n0 = blockIdx.y * 64;
  (void)M;

  // double-buffered B tile: 64 rows x 32 halfs, padded row stride 40 halfs (80B)
  __shared__ __align__(16) _Float16 bsm[2][64 * 40];

  // async staging assignment: thread -> (row, 16B chunk) of the tile
  const int srow = w * 8 + (l >> 2);          // 0..63
  const int scol = (l & 3) * 8;               // 0,8,16,24 halfs
  const _Float16* gsrc  = WT + (size_t)(n0 + srow) * K + scol;
  _Float16* ldst0 = &bsm[0][srow * 40 + scol];
  _Float16* ldst1 = &bsm[1][srow * 40 + scol];

  const _Float16* arow = A + (size_t)(m0 + ln) * K + hw * 8;

  v8f acc[4];
#pragma unroll
  for (int j = 0; j < 4; ++j)
#pragma unroll
    for (int i = 0; i < 8; ++i) acc[j][i] = 0.f;

  const int nIt = K / 32;
  async_copy_b128(ldst0, gsrc);               // tile 0 -> buf 0

  for (int it = 0; it < nIt - 1; ++it) {
    // stream tile it+1 while computing tile it
    async_copy_b128(((it + 1) & 1) ? ldst1 : ldst0, gsrc + (size_t)(it + 1) * 32);
    wait_async1();                            // tile it complete (ours)
    __syncthreads();                          // everyone's tile-it writes visible

    __builtin_prefetch(arow + it * 32 + 512, 0, 1);   // A stream lookahead
    v16h a = frag_ld(arow + (size_t)it * 32);
    const _Float16* bbase = bsm[it & 1];
#pragma unroll
    for (int j = 0; j < 4; ++j) {
      const _Float16* bp = bbase + (j * 16 + ln) * 40 + hw * 8;
      v8h lo = *(const v8h*)bp;
      v8h hi = *(const v8h*)(bp + 16);
      v16h b = __builtin_shufflevector(lo, hi, 0,1,2,3,4,5,6,7,8,9,10,11,12,13,14,15);
      acc[j] = wmma16(a, b, acc[j]);
    }
    __syncthreads();                          // done reading buf[it&1]
  }

  {                                           // last K-step (peeled: literal waits)
    const int it = nIt - 1;
    wait_async0();
    __syncthreads();
    v16h a = frag_ld(arow + (size_t)it * 32);
    const _Float16* bbase = bsm[it & 1];
#pragma unroll
    for (int j = 0; j < 4; ++j) {
      const _Float16* bp = bbase + (j * 16 + ln) * 40 + hw * 8;
      v8h lo = *(const v8h*)bp;
      v8h hi = *(const v8h*)(bp + 16);
      v16h b = __builtin_shufflevector(lo, hi, 0,1,2,3,4,5,6,7,8,9,10,11,12,13,14,15);
      acc[j] = wmma16(a, b, acc[j]);
    }
  }

  // D layout: lane owns column (n0 + j*16 + ln), rows m0 + hw*8 + r, r=0..7
  if (MODE == 0) {
    _Float16* C = (_Float16*)Cout;
#pragma unroll
    for (int j = 0; j < 4; ++j)
#pragma unroll
      for (int r = 0; r < 8; ++r)
        C[(size_t)(m0 + hw * 8 + r) * N + n0 + j * 16 + ln] =
            (_Float16)(acc[j][r] * scale);
  } else if (MODE == 1) {
    _Float16* C = (_Float16*)Cout;
    const int mbase = m0 + hw * 8;
    const int bb = mbase >> 11;               // TSEQ_=2048 rows per batch
    const int t  = mbase & (TSEQ_ - 1);
#pragma unroll
    for (int j = 0; j < 4; ++j) {
      const int n = n0 + j * 16 + ln;
      v8h pk;
#pragma unroll
      for (int r = 0; r < 8; ++r) pk[r] = (_Float16)(acc[j][r] * scale);
      *(v8h*)(C + ((size_t)(bb * N + n)) * TSEQ_ + t) = pk;   // contiguous along t
    }
  } else {
    float* C = (float*)Cout;
#pragma unroll
    for (int j = 0; j < 4; ++j)
#pragma unroll
      for (int r = 0; r < 8; ++r)
        C[(size_t)(m0 + hw * 8 + r) * N + n0 + j * 16 + ln] = acc[j][r] * scale;
  }
}

// ---------------------------------------------------------------------------
// Flash attention, one wave (32 thr) per (b, h, 16-query tile).
// q pre-scaled by d^-0.5 * log2(e) -> softmax via exp2 (v_exp_f32).
// S via wmma (Q as A frag, K rows as B frag: contiguous loads). P reshaped
// C/D-layout -> A-layout through LDS; V consumed from vT[d][T] so its
// B fragments are contiguous loads too.
// ---------------------------------------------------------------------------
__global__ __launch_bounds__(32) void attn_wmma(const _Float16* __restrict__ q,
                                                const _Float16* __restrict__ k,
                                                const _Float16* __restrict__ vT,
                                                _Float16* __restrict__ o) {
  const int l  = threadIdx.x;
  const int ln = l & 15;
  const int hw = l >> 4;
  const int qt = blockIdx.x;
  const int h  = blockIdx.y;
  const int b  = blockIdx.z;

  const _Float16* qb = q  + ((size_t)b * TSEQ_) * INNER_ + h * DHEAD_;
  const _Float16* kb = k  + ((size_t)b * TSEQ_) * INNER_ + h * DHEAD_;
  const _Float16* vb = vT + ((size_t)(b * HEADS_ + h)) * DHEAD_ * TSEQ_; // [d][T]

  // Q tile 16x64 -> two A fragments (K=0..31, 32..63)
  v16h qa[2];
  {
    const _Float16* qr = qb + (size_t)(qt * 16 + ln) * INNER_ + hw * 8;
    qa[0] = frag_ld(qr);
    qa[1] = frag_ld(qr + 32);
  }

  v8f acc[4];
#pragma unroll
  for (int j = 0; j < 4; ++j)
#pragma unroll
    for (int i = 0; i < 8; ++i) acc[j][i] = 0.f;

  float mrow[8], lrow[8];
#pragma unroll
  for (int r = 0; r < 8; ++r) { mrow[r] = -1e30f; lrow[r] = 0.f; }

  __shared__ __align__(16) _Float16 pls[16 * 80];   // 16 rows x 64 keys, stride 80

  for (int kk = 0; kk < TSEQ_; kk += 64) {
    // ---- S = Q K^T for 64 keys: 4 N-tiles x 2 K-deep WMMAs
    v8f S[4];
#pragma unroll
    for (int j = 0; j < 4; ++j) {
#pragma unroll
      for (int i = 0; i < 8; ++i) S[j][i] = 0.f;
      const _Float16* kr = kb + (size_t)(kk + j * 16 + ln) * INNER_ + hw * 8;
      S[j] = wmma16(qa[0], frag_ld(kr), S[j]);
      S[j] = wmma16(qa[1], frag_ld(kr + 32), S[j]);
    }

    // ---- online softmax (rows live across 16-lane groups in C/D layout)
    float alpha[8];
#pragma unroll
    for (int r = 0; r < 8; ++r) {
      float mx = fmaxf(fmaxf(S[0][r], S[1][r]), fmaxf(S[2][r], S[3][r]));
      mx = fmaxf(mx, __shfl_xor(mx, 1, 16));
      mx = fmaxf(mx, __shfl_xor(mx, 2, 16));
      mx = fmaxf(mx, __shfl_xor(mx, 4, 16));
      mx = fmaxf(mx, __shfl_xor(mx, 8, 16));
      const float mn = fmaxf(mrow[r], mx);
      alpha[r] = exp2f(mrow[r] - mn);
      mrow[r] = mn;
    }

    float ps[8];
#pragma unroll
    for (int r = 0; r < 8; ++r) ps[r] = 0.f;
#pragma unroll
    for (int j = 0; j < 4; ++j)
#pragma unroll
      for (int r = 0; r < 8; ++r) {
        const float p = exp2f(S[j][r] - mrow[r]);
        ps[r] += p;
        pls[(hw * 8 + r) * 80 + j * 16 + ln] = (_Float16)p;
      }

#pragma unroll
    for (int r = 0; r < 8; ++r) {
      float s = ps[r];
      s += __shfl_xor(s, 1, 16);
      s += __shfl_xor(s, 2, 16);
      s += __shfl_xor(s, 4, 16);
      s += __shfl_xor(s, 8, 16);
      lrow[r] = lrow[r] * alpha[r] + s;
#pragma unroll
      for (int j = 0; j < 4; ++j) acc[j][r] *= alpha[r];
    }

    __syncthreads();   // single-wave WG: barrier -> NOP, keeps ds waits/ordering

    // ---- reload P in A-fragment layout from LDS
    v16h pa[2];
#pragma unroll
    for (int kf = 0; kf < 2; ++kf) {
      const _Float16* pp = &pls[ln * 80 + kf * 32 + hw * 8];
      v8h lo = *(const v8h*)pp;
      v8h hi = *(const v8h*)(pp + 16);
      pa[kf] = __builtin_shufflevector(lo, hi, 0,1,2,3,4,5,6,7,8,9,10,11,12,13,14,15);
    }

    // ---- O += P @ V   (V from vT: B fragment loads contiguous along T)
#pragma unroll
    for (int j = 0; j < 4; ++j) {
      const _Float16* vr = vb + (size_t)(j * 16 + ln) * TSEQ_ + kk + hw * 8;
      acc[j] = wmma16(pa[0], frag_ld(vr), acc[j]);
      acc[j] = wmma16(pa[1], frag_ld(vr + 32), acc[j]);
    }

    __syncthreads();
  }

  // ---- epilogue: normalize and store f16 row-major [T, h*d]
  _Float16* ob = o + ((size_t)b * TSEQ_) * INNER_ + h * DHEAD_;
#pragma unroll
  for (int j = 0; j < 4; ++j)
#pragma unroll
    for (int r = 0; r < 8; ++r)
      ob[(size_t)(qt * 16 + hw * 8 + r) * INNER_ + j * 16 + ln] =
          (_Float16)(acc[j][r] / lrow[r]);
}

// ---------------------------------------------------------------------------
// Host-side launcher
// ---------------------------------------------------------------------------
extern "C" void kernel_launch(void* const* d_in, const int* in_sizes, int n_in,
                              void* d_out, int out_size, void* d_ws, size_t ws_size,
                              hipStream_t stream) {
  (void)in_sizes; (void)n_in; (void)out_size; (void)ws_size;

  const float* x   = (const float*)d_in[0];   // [4,2048,1024]
  const float* y   = (const float*)d_in[1];   // [4,2048,1024]
  const float* lns = (const float*)d_in[2];   // [1024]
  const float* lnb = (const float*)d_in[3];   // [1024]
  const float* Wq  = (const float*)d_in[4];   // [1024,512]
  const float* Wkv = (const float*)d_in[5];   // [1024,1024]
  const float* Wo  = (const float*)d_in[6];   // [512,1024]
  float* out = (float*)d_out;                 // [4,2048,1024] f32

  // workspace layout (f16 elements)
  _Float16* ws  = (_Float16*)d_ws;
  _Float16* xn  = ws;                          // 8192*1024
  _Float16* yh  = xn  + (size_t)ROWS_ * DIM_;  // 8192*1024
  _Float16* qh  = yh  + (size_t)ROWS_ * DIM_;  // 8192*512
  _Float16* kh  = qh  + (size_t)ROWS_ * INNER_;
  _Float16* vT  = kh  + (size_t)ROWS_ * INNER_; // [b][h][d][T]
  _Float16* oh  = vT  + (size_t)ROWS_ * INNER_;
  _Float16* WqT = oh  + (size_t)ROWS_ * INNER_; // [512][1024]
  _Float16* WkT = WqT + (size_t)INNER_ * DIM_;  // [512][1024]
  _Float16* WvT = WkT + (size_t)INNER_ * DIM_;  // [512][1024]
  _Float16* WoT = WvT + (size_t)INNER_ * DIM_;  // [1024][512]

  // 1) weight prep (transpose + f16)
  wt_trans<<<(DIM_ * INNER_ + 255) / 256, 256, 0, stream>>>(Wq,  WqT, DIM_,   INNER_, INNER_,     0);
  wt_trans<<<(DIM_ * INNER_ + 255) / 256, 256, 0, stream>>>(Wkv, WkT, DIM_,   INNER_, 2 * INNER_, 0);
  wt_trans<<<(DIM_ * INNER_ + 255) / 256, 256, 0, stream>>>(Wkv, WvT, DIM_,   INNER_, 2 * INNER_, INNER_);
  wt_trans<<<(INNER_ * DIM_ + 255) / 256, 256, 0, stream>>>(Wo,  WoT, INNER_, DIM_,   DIM_,       0);

  // 2) layernorm(x) -> f16 ; y -> f16
  ln_f16<<<ROWS_, 256, 0, stream>>>(x, lns, lnb, xn);
  conv_f16<<<(ROWS_ * DIM_ / 4 + 255) / 256, 256, 0, stream>>>(y, yh, ROWS_ * DIM_ / 4);

  // 3) projections (q folds d^-0.5 * log2(e) for base-2 softmax)
  const float qscale = 0.125f * 1.44269504088896341f;
  dim3 g512(ROWS_ / 128, INNER_ / 64);
  gemm_wmma<0><<<g512, 256, 0, stream>>>(xn, WqT, (void*)qh, ROWS_, INNER_, DIM_, qscale);
  gemm_wmma<0><<<g512, 256, 0, stream>>>(yh, WkT, (void*)kh, ROWS_, INNER_, DIM_, 1.f);
  gemm_wmma<1><<<g512, 256, 0, stream>>>(yh, WvT, (void*)vT, ROWS_, INNER_, DIM_, 1.f);

  // 4) flash attention
  dim3 ga(TSEQ_ / 16, HEADS_, BATCH_);
  attn_wmma<<<ga, 32, 0, stream>>>(qh, kh, vT, oh);

  // 5) output projection -> f32 d_out
  dim3 go(ROWS_ / 128, DIM_ / 64);
  gemm_wmma<2><<<go, 256, 0, stream>>>(oh, WoT, (void*)out, ROWS_, DIM_, INNER_, 1.f);
}
